// StandardHead_1382979469657
// MI455X (gfx1250) — compile-verified, matrix operand order checked by reading, not run on hardware
//
#include <hip/hip_runtime.h>

typedef __attribute__((ext_vector_type(16))) _Float16 v16h;
typedef __attribute__((ext_vector_type(8)))  _Float16 v8h;
typedef __attribute__((ext_vector_type(4)))  _Float16 v4h;
typedef __attribute__((ext_vector_type(8)))  float    v8f;

constexpr int Bb = 4, Tt = 2048, Cc = 1024, HS = 64;

// Load a 16-element f16 WMMA fragment whose per-lane elements are two
// contiguous 8-half runs: [h8 + 0..7] and [16 + h8 + 0..7] from rowp.
// (A-frag: rowp = row(lane%16) base; B-frag: rowp = col(lane%16) base of the
// transposed-source operand.)  h8 = 8*(lane>=16).
__device__ inline v16h frag_from_contig(const _Float16* rowp, int h8) {
  v8h lo = *(const v8h*)(rowp + h8);
  v8h hi = *(const v8h*)(rowp + 16 + h8);
  v16h r;
#pragma unroll
  for (int i = 0; i < 8; ++i) { r[i] = lo[i]; r[i + 8] = hi[i]; }
  return r;
}

// ---------------- Phase 1: fused QKV projection (x read once) ---------------
// grid.x = B*T/16, block = 128 (4 waves).  Each block: 16 tokens, all 64 HS,
// all three outputs.  Wave w owns HS tile w.  K-loop over C in chunks of 32.
// W chunks are staged in LDS TRANSPOSED (lw[m][hs*32 + c]) so that both A and
// B fragments are read with contiguous ds_load_b128s.
__global__ __launch_bounds__(128)
void qkv_proj_kernel(const float* __restrict__ x,
                     const float* __restrict__ Wq,
                     const float* __restrict__ Wk,
                     const float* __restrict__ Wv,
                     _Float16* __restrict__ qh,   // [B*T][HS] f16
                     _Float16* __restrict__ kh,   // [B*T][HS] f16
                     _Float16* __restrict__ vT)   // [B][HS][T] f16 (transposed)
{
  __shared__ _Float16 lw[3][64 * 32];  // W chunk, transposed, f16, 12 KB
  __shared__ _Float16 lx[16 * 32];     // x chunk, f16, 1 KB

  const int tid  = threadIdx.x;
  const int lane = tid & 31;
  const int wv   = tid >> 5;           // 0..3 = HS tile
  const int col  = lane & 15;
  const int h8   = (lane >> 4) << 3;   // 0 or 8
  const int tok0 = blockIdx.x * 16;

  // staging roles: x -> 4 contiguous elems/thread; W -> one (hs, 16-c run)
  const int whs = tid & 63;            // 0..63
  const int wc0 = (tid >> 6) << 4;     // 0 or 16

  const float* Wm[3] = {Wq, Wk, Wv};
  v8f accq = {}, acck = {}, accv = {};

  for (int ch = 0; ch < Cc / 32; ++ch) {
    __syncthreads();  // previous iteration's readers done
    {   // stage x chunk 16x32 f32 -> f16 (4 elems/thread, one b64 LDS store)
      int flat = tid * 4;
      int row = flat >> 5, c = flat & 31;
      const float4 xv = *(const float4*)(x + (size_t)(tok0 + row) * Cc + ch * 32 + c);
      v4h h; h[0] = (_Float16)xv.x; h[1] = (_Float16)xv.y;
             h[2] = (_Float16)xv.z; h[3] = (_Float16)xv.w;
      *(v4h*)&lx[flat] = h;
    }
#pragma unroll
    for (int m = 0; m < 3; ++m) {   // stage W chunk transposed: lw[hs*32 + c]
      v8h t0, t1;
#pragma unroll
      for (int i = 0; i < 8; ++i) {
        t0[i] = (_Float16)Wm[m][(size_t)(ch * 32 + wc0 + i) * HS + whs];
        t1[i] = (_Float16)Wm[m][(size_t)(ch * 32 + wc0 + 8 + i) * HS + whs];
      }
      *(v8h*)&lw[m][whs * 32 + wc0]     = t0;
      *(v8h*)&lw[m][whs * 32 + wc0 + 8] = t1;
    }
    __syncthreads();

    v16h af = frag_from_contig(&lx[col * 32], h8);              // A: tokens x C
    v16h bq = frag_from_contig(&lw[0][(wv * 16 + col) * 32], h8); // B: C x HS
    v16h bk = frag_from_contig(&lw[1][(wv * 16 + col) * 32], h8);
    v16h bv = frag_from_contig(&lw[2][(wv * 16 + col) * 32], h8);

    accq = __builtin_amdgcn_wmma_f32_16x16x32_f16(false, af, false, bq, (short)0, accq, false, false);
    acck = __builtin_amdgcn_wmma_f32_16x16x32_f16(false, af, false, bk, (short)0, acck, false, false);
    accv = __builtin_amdgcn_wmma_f32_16x16x32_f16(false, af, false, bv, (short)0, accv, false, false);
  }

#pragma unroll
  for (int r = 0; r < 8; ++r) {        // D layout: row = r + h8, col = lane%16
    int tok = tok0 + r + h8;
    int hs  = wv * 16 + col;
    qh[(size_t)tok * HS + hs] = (_Float16)accq[r];
    kh[(size_t)tok * HS + hs] = (_Float16)acck[r];
    int b = tok >> 11, tl = tok & (Tt - 1);
    vT[((size_t)b * HS + hs) * Tt + tl] = (_Float16)accv[r];
  }
}

// ---------------- Phase 2: flash attention ----------------------------------
// grid = (T/64, B), block = 128 (4 waves).  Wave w owns 16 query rows; key
// blocks of 64 keys; online softmax; P bounced through per-wave LDS tile.
__global__ __launch_bounds__(128)
void attn_kernel(const _Float16* __restrict__ qh,
                 const _Float16* __restrict__ kh,
                 const _Float16* __restrict__ vT,
                 float* __restrict__ out)
{
  __shared__ _Float16 P[4][16 * 64];   // per-wave private P tile, 8 KB total

  const int tid  = threadIdx.x;
  const int lane = tid & 31;
  const int wv   = tid >> 5;
  const int col  = lane & 15;
  const int h8   = (lane >> 4) << 3;
  const int b    = blockIdx.y;
  const int qr0  = blockIdx.x * 64 + wv * 16;   // wave's first query row
  const size_t qbase = (size_t)b * Tt;

  const _Float16* qrow = qh + (qbase + qr0 + col) * HS;
  v16h qf0 = frag_from_contig(qrow, h8);        // HS chunk 0..31
  v16h qf1 = frag_from_contig(qrow + 32, h8);   // HS chunk 32..63

  v8f O0 = {}, O1 = {}, O2 = {}, O3 = {};
  float mrow[8], lrow[8];
#pragma unroll
  for (int r = 0; r < 8; ++r) { mrow[r] = -__builtin_inff(); lrow[r] = 0.f; }

  const float scale = 0.03125f;  // C^-0.5 = 1/32

  const int kb_last = qr0;       // prefetch bound (= last key block base only for wave 0)
  for (int kb = 0; kb <= qr0 + 15; kb += 64) {
    // prefetch next key block's K / V^T rows (global_prefetch_b8)
    if (kb < kb_last) {
      __builtin_prefetch(kh + (qbase + kb + 64 + col) * HS, 0, 1);
      __builtin_prefetch(vT + ((size_t)b * HS + wv * 16 + col) * Tt + kb + 64, 0, 1);
    }
    // S = Q @ K^T  (4 key tiles, K-dim = HS = 64 -> 2 WMMAs each)
    v8f S[4];
#pragma unroll
    for (int nt = 0; nt < 4; ++nt) {
      const _Float16* krow = kh + (qbase + kb + nt * 16 + col) * HS;
      v16h kf0 = frag_from_contig(krow, h8);
      v16h kf1 = frag_from_contig(krow + 32, h8);
      v8f s = {};
      s = __builtin_amdgcn_wmma_f32_16x16x32_f16(false, qf0, false, kf0, (short)0, s, false, false);
      s = __builtin_amdgcn_wmma_f32_16x16x32_f16(false, qf1, false, kf1, (short)0, s, false, false);
      S[nt] = s;
    }
    // scale + causal mask + row max (xor-reduce across the 16-lane half)
    float nm[8];
#pragma unroll
    for (int r = 0; r < 8; ++r) {
      int qg = qr0 + r + h8;
      float mx = -__builtin_inff();
#pragma unroll
      for (int nt = 0; nt < 4; ++nt) {
        int kcol = kb + nt * 16 + col;
        float sv = S[nt][r] * scale;
        if (kcol > qg) sv = -__builtin_inff();
        S[nt][r] = sv;
        mx = fmaxf(mx, sv);
      }
#pragma unroll
      for (int off = 1; off < 16; off <<= 1)
        mx = fmaxf(mx, __shfl_xor(mx, off, 32));
      nm[r] = fmaxf(mrow[r], mx);
    }
    // rescale accumulators, P = exp(S - m), row sums
#pragma unroll
    for (int r = 0; r < 8; ++r) {
      float corr = __expf(mrow[r] - nm[r]);   // first block: exp(-inf) = 0
      lrow[r] *= corr;
      O0[r] *= corr; O1[r] *= corr; O2[r] *= corr; O3[r] *= corr;
      float ps = 0.f;
#pragma unroll
      for (int nt = 0; nt < 4; ++nt) {
        float p = __expf(S[nt][r] - nm[r]);   // masked: exp(-inf) = 0
        P[wv][(r + h8) * 64 + nt * 16 + col] = (_Float16)p;
        ps += p;
      }
#pragma unroll
      for (int off = 1; off < 16; off <<= 1)
        ps += __shfl_xor(ps, off, 32);
      lrow[r] += ps;
      mrow[r] = nm[r];
    }
    // cross-lane LDS store->load hazard within this wave
    asm volatile("s_wait_dscnt 0" ::: "memory");
    // O += P @ V  (K-dim = 64 keys -> 2 chunks of 32; V is transposed)
#pragma unroll
    for (int ch = 0; ch < 2; ++ch) {
      v16h pf = frag_from_contig(&P[wv][col * 64 + ch * 32], h8);
#pragma unroll
      for (int nt = 0; nt < 4; ++nt) {
        const _Float16* vrow = vT + ((size_t)b * HS + nt * 16 + col) * Tt + kb + ch * 32;
        v16h vf = frag_from_contig(vrow, h8);
        v8f* Op = (nt == 0) ? &O0 : (nt == 1) ? &O1 : (nt == 2) ? &O2 : &O3;
        *Op = __builtin_amdgcn_wmma_f32_16x16x32_f16(false, pf, false, vf, (short)0, *Op, false, false);
      }
    }
  }

#pragma unroll
  for (int r = 0; r < 8; ++r) {
    float inv = 1.0f / lrow[r];
    int qg = qr0 + r + h8;
    float* orow = out + (qbase + qg) * (size_t)HS;
    orow[ 0 + col] = O0[r] * inv;
    orow[16 + col] = O1[r] * inv;
    orow[32 + col] = O2[r] * inv;
    orow[48 + col] = O3[r] * inv;
  }
}

extern "C" void kernel_launch(void* const* d_in, const int* in_sizes, int n_in,
                              void* d_out, int out_size, void* d_ws, size_t ws_size,
                              hipStream_t stream) {
  const float* x  = (const float*)d_in[0];
  const float* Wq = (const float*)d_in[1];
  const float* Wk = (const float*)d_in[2];
  const float* Wv = (const float*)d_in[3];
  float* out = (float*)d_out;

  _Float16* qh = (_Float16*)d_ws;                 // 1 MB
  _Float16* kh = qh + (size_t)Bb * Tt * HS;       // 1 MB
  _Float16* vT = kh + (size_t)Bb * Tt * HS;       // 1 MB

  dim3 g1((Bb * Tt) / 16, 1, 1);
  qkv_proj_kernel<<<g1, 128, 0, stream>>>(x, Wq, Wk, Wv, qh, kh, vT);
  dim3 g2(Tt / 64, Bb, 1);
  attn_kernel<<<g2, 128, 0, stream>>>(qh, kh, vT, out);
}
